// DyHGCN_H_38577396253203
// MI455X (gfx1250) — matrix operand order repaired and checked by reading, not compiled, not run on hardware
//
#include <hip/hip_runtime.h>
#include <hip/hip_bf16.h>
#include <math.h>

// ---- problem constants (from reference setup_inputs) ----
#define Vv   50000
#define Dd   64
#define Tt   8
#define Bb   8
#define Ll   128
#define L1   129
#define Ee   200000
#define NLn  2
#define FFf  256
#define NHh  8
#define DHh  8

typedef __attribute__((ext_vector_type(16))) _Float16 v16h;
typedef __attribute__((ext_vector_type(8)))  float    v8f;

// =====================================================================
// 1) dyemb + pos_emb  -> final[1024][64]
//    One block per query (b,l). Scan the t-th edge list, compact matches
//    into LDS, sort (deterministic FP order), recompute xW[src] on the fly.
// =====================================================================
__global__ void k_dyemb(const float* __restrict__ emb, const float* __restrict__ Wg,
                        const float* __restrict__ ew, const float* __restrict__ pos,
                        const int* __restrict__ inSeq, const int* __restrict__ tIdx,
                        const int* __restrict__ esrc, const int* __restrict__ edst,
                        float* __restrict__ finalBuf) {
    __shared__ int list[1024];
    __shared__ int cnt;
    int j = blockIdx.x;           // 0..1023  == b*128 + l
    int b = j >> 7, l = j & 127;
    int t = tIdx[j];
    int node = inSeq[b * L1 + l];
    int tid = threadIdx.x;
    if (tid == 0) cnt = 0;
    __syncthreads();
    const int* dstp = edst + (long long)t * Ee;
    for (int e = tid; e < Ee; e += 256) {
        if (dstp[e] == node) {
            int p = atomicAdd(&cnt, 1);
            if (p < 1024) list[p] = e;
        }
    }
    __syncthreads();
    int m = cnt; if (m > 1024) m = 1024;
    if (tid == 0) {  // insertion sort of matched edge ids -> deterministic sum order
        for (int i = 1; i < m; ++i) {
            int key = list[i]; int p = i - 1;
            while (p >= 0 && list[p] > key) { list[p + 1] = list[p]; --p; }
            list[p + 1] = key;
        }
    }
    __syncthreads();
    if (tid < Dd) {
        float acc = 0.f;
        for (int i = 0; i < m; ++i) {
            int e = list[i];
            float w = ew[(long long)t * Ee + e];
            int src = esrc[(long long)t * Ee + e];
            const float* er = emb + (long long)src * Dd;
            float x = 0.f;
            #pragma unroll 8
            for (int e2 = 0; e2 < Dd; ++e2) x += er[e2] * Wg[e2 * Dd + tid];
            acc += w * x;
        }
        finalBuf[j * Dd + tid] = acc + pos[l * Dd + tid];
    }
}

// =====================================================================
// 2) first attention: score (masked), softmax over q (axis=1), att -> h
// =====================================================================
__global__ void k_score(const float* __restrict__ finalBuf, const float* __restrict__ temb,
                        const int* __restrict__ tIdx, const int* __restrict__ inSeq,
                        float* __restrict__ score) {
    int idx = blockIdx.x * 256 + threadIdx.x;       // b*16384 + q*128 + k
    if (idx >= Bb * Ll * Ll) return;
    int b = idx >> 14, q = (idx >> 7) & 127, k = idx & 127;
    float s;
    if (k > q || inSeq[b * L1 + q] == 0) {
        s = -4294967295.0f;                          // NEG = -2^32+1
    } else {
        int t = tIdx[b * Ll + k];
        const float* f  = finalBuf + (b * Ll + q) * Dd;
        const float* te = temb + t * Dd;
        s = 0.f;
        #pragma unroll 8
        for (int e = 0; e < Dd; ++e) s += f[e] * te[e];
        s /= 8.000001f;                              // sqrt(64)+1e-6
    }
    score[idx] = s;
}

__global__ void k_softmax_q(float* __restrict__ score) {
    __shared__ float red[128];
    int b = blockIdx.x >> 7, k = blockIdx.x & 127;   // column (b,k)
    int q = threadIdx.x;                             // 128 threads
    float v = score[(b * Ll + q) * Ll + k];
    red[q] = v; __syncthreads();
    for (int s = 64; s > 0; s >>= 1) { if (q < s) red[q] = fmaxf(red[q], red[q + s]); __syncthreads(); }
    float mx = red[0]; __syncthreads();
    float e = __expf(v - mx);
    red[q] = e; __syncthreads();
    for (int s = 64; s > 0; s >>= 1) { if (q < s) red[q] += red[q + s]; __syncthreads(); }
    float sum = red[0];
    score[(b * Ll + q) * Ll + k] = e / sum;
}

__global__ void k_att0(const float* __restrict__ alpha, const float* __restrict__ finalBuf,
                       float* __restrict__ h) {
    int idx = blockIdx.x * 256 + threadIdx.x;        // b*8192 + q*64 + d
    int d = idx & 63, q = (idx >> 6) & 127, b = idx >> 13;
    const float* ar = alpha + (b * Ll + q) * Ll;
    float acc = 0.f;
    #pragma unroll 4
    for (int k = 0; k < Ll; ++k) acc += ar[k] * finalBuf[(b * Ll + k) * Dd + d];
    h[idx] = acc;
}

// =====================================================================
// 3) transformer layer kernels (d=64, 8 heads x 8, FF=256)
// =====================================================================
__global__ void k_qkv(const float* __restrict__ h, const float* __restrict__ Wq,
                      const float* __restrict__ Wk, const float* __restrict__ Wv,
                      float* __restrict__ qkv, int layer) {
    int idx = blockIdx.x * 256 + threadIdx.x;        // which*65536 + row*64 + col
    int which = idx >> 16, row = (idx >> 6) & 1023, col = idx & 63;
    const float* W = (which == 0 ? Wq : which == 1 ? Wk : Wv) + layer * Dd * Dd;
    const float* hr = h + row * Dd;
    float acc = 0.f;
    #pragma unroll 8
    for (int e = 0; e < Dd; ++e) acc += hr[e] * W[e * Dd + col];
    qkv[idx] = acc;
}

__global__ void k_attn(const float* __restrict__ qkv, const int* __restrict__ inSeq,
                       float* __restrict__ obuf) {
    __shared__ float ksh[Ll][DHh];
    __shared__ float vsh[Ll][DHh];
    __shared__ int   pads[Ll];
    int b = blockIdx.x >> 3, hh = blockIdx.x & 7;
    int tid = threadIdx.x;                           // 128 threads, thread = q row
    const float* qb = qkv;
    const float* kb = qkv + 1024 * Dd;
    const float* vb = qkv + 2 * 1024 * Dd;
    for (int i = tid; i < Ll * DHh; i += 128) {
        int kr = i >> 3, e = i & 7;
        ksh[kr][e] = kb[(b * Ll + kr) * Dd + hh * DHh + e];
        vsh[kr][e] = vb[(b * Ll + kr) * Dd + hh * DHh + e];
    }
    pads[tid] = (inSeq[b * L1 + tid] == 0);
    __syncthreads();
    int q = tid;
    float qv[DHh];
    #pragma unroll
    for (int e = 0; e < DHh; ++e) qv[e] = qb[(b * Ll + q) * Dd + hh * DHh + e];
    const float scale = 0.35355339059327373f;        // 1/sqrt(8)
    float mx = -3.4e38f;
    for (int k = 0; k < Ll; ++k) {
        float s;
        if (k > q || pads[k]) s = -1.0e9f;
        else {
            s = 0.f;
            #pragma unroll
            for (int e = 0; e < DHh; ++e) s += qv[e] * ksh[k][e];
            s *= scale;
        }
        mx = fmaxf(mx, s);
    }
    float sum = 0.f, oa[DHh] = {0, 0, 0, 0, 0, 0, 0, 0};
    for (int k = 0; k < Ll; ++k) {
        float s;
        if (k > q || pads[k]) s = -1.0e9f;
        else {
            s = 0.f;
            #pragma unroll
            for (int e = 0; e < DHh; ++e) s += qv[e] * ksh[k][e];
            s *= scale;
        }
        float ee = __expf(s - mx);
        sum += ee;
        #pragma unroll
        for (int e = 0; e < DHh; ++e) oa[e] += ee * vsh[k][e];
    }
    float inv = 1.0f / sum;
    #pragma unroll
    for (int e = 0; e < DHh; ++e)
        obuf[(b * Ll + q) * Dd + hh * DHh + e] = oa[e] * inv;
}

__global__ void k_proj_ln(const float* __restrict__ obuf, const float* __restrict__ Wo,
                          const float* __restrict__ g, const float* __restrict__ be,
                          float* __restrict__ h, int layer) {
    __shared__ float red[Dd];
    int row = blockIdx.x, d = threadIdx.x;           // 64 threads per row
    const float* W = Wo + layer * Dd * Dd;
    const float* orow = obuf + row * Dd;
    float s = 0.f;
    #pragma unroll 8
    for (int e = 0; e < Dd; ++e) s += orow[e] * W[e * Dd + d];
    float x = h[row * Dd + d] + s;
    red[d] = x; __syncthreads();
    for (int st = 32; st > 0; st >>= 1) { if (d < st) red[d] += red[d + st]; __syncthreads(); }
    float mean = red[0] * (1.0f / Dd); __syncthreads();
    float c = x - mean;
    red[d] = c * c; __syncthreads();
    for (int st = 32; st > 0; st >>= 1) { if (d < st) red[d] += red[d + st]; __syncthreads(); }
    float var = red[0] * (1.0f / Dd);
    h[row * Dd + d] = c * rsqrtf(var + 1e-6f) * g[layer * Dd + d] + be[layer * Dd + d];
}

__global__ void k_ff1(const float* __restrict__ h, const float* __restrict__ W1,
                      float* __restrict__ f1, int layer) {
    int idx = blockIdx.x * 256 + threadIdx.x;        // row*256 + jc
    int row = idx >> 8, jc = idx & 255;
    const float* W = W1 + layer * Dd * FFf;
    const float* hr = h + row * Dd;
    float acc = 0.f;
    #pragma unroll 8
    for (int e = 0; e < Dd; ++e) acc += hr[e] * W[e * FFf + jc];
    f1[idx] = fmaxf(acc, 0.f);
}

__global__ void k_ff2_ln(const float* __restrict__ f1, const float* __restrict__ W2,
                         const float* __restrict__ g, const float* __restrict__ be,
                         float* __restrict__ h, int layer) {
    __shared__ float red[Dd];
    int row = blockIdx.x, d = threadIdx.x;
    const float* W = W2 + layer * FFf * Dd;
    const float* fr = f1 + row * FFf;
    float s = 0.f;
    #pragma unroll 8
    for (int j = 0; j < FFf; ++j) s += fr[j] * W[j * Dd + d];
    float x = h[row * Dd + d] + s;
    red[d] = x; __syncthreads();
    for (int st = 32; st > 0; st >>= 1) { if (d < st) red[d] += red[d + st]; __syncthreads(); }
    float mean = red[0] * (1.0f / Dd); __syncthreads();
    float c = x - mean;
    red[d] = c * c; __syncthreads();
    for (int st = 32; st > 0; st >>= 1) { if (d < st) red[d] += red[d + st]; __syncthreads(); }
    float var = red[0] * (1.0f / Dd);
    h[row * Dd + d] = c * rsqrtf(var + 1e-6f) * g[layer * Dd + d] + be[layer * Dd + d];
}

// =====================================================================
// 4) f32 -> f16 conversion
// =====================================================================
__global__ void k_cvt_f16(const float* __restrict__ src, _Float16* __restrict__ dst, int n) {
    int i = blockIdx.x * 256 + threadIdx.x;
    if (i < n) dst[i] = (_Float16)src[i];
}

// =====================================================================
// 5) logits GEMM: out[1024 x 50000] = hF[1024x64] @ wF[50000x64]^T + bias
//    WMMA f32_16x16x32_f16; one 16x16 tile per wave, K=64 = 2 chained WMMAs.
// =====================================================================
__global__ void k_logits(const _Float16* __restrict__ hF, const _Float16* __restrict__ wF,
                         const float* __restrict__ bias, float* __restrict__ out) {
    int tid  = threadIdx.x;
    int wid  = tid >> 5;
    int lane = tid & 31;
    int ntile = blockIdx.x * 8 + wid;
    if (ntile >= Vv / 16) return;                    // wave-uniform tail skip (3125 tiles)
    int hs  = lane >> 4;                             // lane half (0: lanes 0-15, 1: 16-31)
    int l16 = lane & 15;
    const _Float16* arow = hF + (blockIdx.y * 16 + l16) * Dd;   // A: M row
    const _Float16* brow = wF + (ntile * 16 + l16) * Dd;        // B: column n = W_out row
    v8f c = {0, 0, 0, 0, 0, 0, 0, 0};
    #pragma unroll
    for (int kb = 0; kb < 2; ++kb) {
        v16h a, b;
        // A 16x32 f16 layout: lane half 0 -> K {0..7,16..23}; half 1 -> K {8..15,24..31}
        const _Float16* ap0 = arow + 32 * kb + 8 * hs;
        const _Float16* ap1 = arow + 32 * kb + 16 + 8 * hs;
        // B 32x16 f16 layout: lane half 0 -> K {0..7,16..23}; half 1 -> K {8..15,24..31}
        const _Float16* bp  = brow + 32 * kb + 8 * hs;
        #pragma unroll
        for (int i = 0; i < 8; ++i) {
            a[i]     = ap0[i];
            a[8 + i] = ap1[i];
            b[i]     = bp[i];
            b[8 + i] = bp[16 + i];
        }
        c = __builtin_amdgcn_wmma_f32_16x16x32_f16(false, a, false, b, (short)0, c,
                                                   false, false);
    }
    int ncol = ntile * 16 + l16;
    float bv = bias[ncol];
    long long base = (long long)(blockIdx.y * 16) * Vv + ncol;
    #pragma unroll
    for (int j = 0; j < 8; ++j) {                    // D: VGPR j -> M = j + 8*hs, N = l16
        int m = j + 8 * hs;
        out[base + (long long)m * Vv] = c[j] + bv;
    }
}

// =====================================================================
// 6) -inf scatter: out[b,q,0] and out[b,q,inp[b,k<=q]] = -inf
// =====================================================================
__global__ void k_mask(const int* __restrict__ inSeq, float* __restrict__ out) {
    int j = blockIdx.x;                              // 0..1023 == b*128 + q
    int b = j >> 7, q = j & 127;
    int k = threadIdx.x;                             // 128 threads
    long long rowb = (long long)j * Vv;
    if (k == 0) out[rowb] = -INFINITY;
    if (k <= q) out[rowb + inSeq[b * L1 + k]] = -INFINITY;
}

// =====================================================================
extern "C" void kernel_launch(void* const* d_in, const int* in_sizes, int n_in,
                              void* d_out, int out_size, void* d_ws, size_t ws_size,
                              hipStream_t stream) {
    const float* embedding = (const float*)d_in[0];
    const float* W_gcn     = (const float*)d_in[1];
    const float* edge_w    = (const float*)d_in[2];
    const float* pos_emb   = (const float*)d_in[3];
    const float* time_emb  = (const float*)d_in[4];
    const float* Wq        = (const float*)d_in[5];
    const float* Wk        = (const float*)d_in[6];
    const float* Wv        = (const float*)d_in[7];
    const float* Wo        = (const float*)d_in[8];
    const float* W1        = (const float*)d_in[9];
    const float* W2        = (const float*)d_in[10];
    const float* ln1_g     = (const float*)d_in[11];
    const float* ln1_b     = (const float*)d_in[12];
    const float* ln2_g     = (const float*)d_in[13];
    const float* ln2_b     = (const float*)d_in[14];
    const float* W_out     = (const float*)d_in[15];
    const float* b_out     = (const float*)d_in[16];
    const int*   input_seq = (const int*)d_in[17];
    const int*   time_idx  = (const int*)d_in[18];
    const int*   edge_src  = (const int*)d_in[19];
    const int*   edge_dst  = (const int*)d_in[20];
    float* out = (float*)d_out;

    // workspace carve (~9.6 MB total)
    char* w = (char*)d_ws;
    float*    finalB = (float*)w;     w += (size_t)1024 * Dd * 4;        // 256 KB
    float*    scoreB = (float*)w;     w += (size_t)Bb * Ll * Ll * 4;     // 512 KB
    float*    hB     = (float*)w;     w += (size_t)1024 * Dd * 4;        // 256 KB
    float*    qkvB   = (float*)w;     w += (size_t)3 * 1024 * Dd * 4;    // 768 KB
    float*    oB     = (float*)w;     w += (size_t)1024 * Dd * 4;        // 256 KB
    float*    f1B    = (float*)w;     w += (size_t)1024 * FFf * 4;       // 1 MB
    _Float16* wF     = (_Float16*)w;  w += (size_t)Vv * Dd * 2;          // 6.4 MB
    _Float16* hF     = (_Float16*)w;                                     // 128 KB

    // W_out -> f16 (independent, kick off first)
    k_cvt_f16<<<(Vv * Dd + 255) / 256, 256, 0, stream>>>(W_out, wF, Vv * Dd);

    // GCN gather (sparse, deterministic) + pos_emb
    k_dyemb<<<1024, 256, 0, stream>>>(embedding, W_gcn, edge_w, pos_emb,
                                      input_seq, time_idx, edge_src, edge_dst, finalB);

    // first attention (softmax over q)
    k_score<<<(Bb * Ll * Ll + 255) / 256, 256, 0, stream>>>(finalB, time_emb, time_idx,
                                                            input_seq, scoreB);
    k_softmax_q<<<Bb * Ll, 128, 0, stream>>>(scoreB);
    k_att0<<<(1024 * Dd + 255) / 256, 256, 0, stream>>>(scoreB, finalB, hB);

    // transformer layers
    for (int l = 0; l < NLn; ++l) {
        k_qkv<<<(3 * 1024 * Dd + 255) / 256, 256, 0, stream>>>(hB, Wq, Wk, Wv, qkvB, l);
        k_attn<<<Bb * NHh, 128, 0, stream>>>(qkvB, input_seq, oB);
        k_proj_ln<<<1024, Dd, 0, stream>>>(oB, Wo, ln1_g, ln1_b, hB, l);
        k_ff1<<<(1024 * FFf + 255) / 256, 256, 0, stream>>>(hB, W1, f1B, l);
        k_ff2_ln<<<1024, Dd, 0, stream>>>(f1B, W2, ln2_g, ln2_b, hB, l);
    }

    // h -> f16, WMMA vocab GEMM, -inf scatter
    k_cvt_f16<<<(1024 * Dd + 255) / 256, 256, 0, stream>>>(hB, hF, 1024 * Dd);
    dim3 g((Vv / 16 + 7) / 8, 1024 / 16);            // (391, 64) blocks x 256 thr
    k_logits<<<g, 256, 0, stream>>>(hF, wF, b_out, out);
    k_mask<<<1024, 128, 0, stream>>>(input_seq, out);
}
